// TransformerEncoderLayer_49091476194111
// MI455X (gfx1250) — matrix operand order repaired and checked by reading, not compiled, not run on hardware
//
#include <hip/hip_runtime.h>
#include <stddef.h>

// ---------------- CDNA5 WMMA types ----------------
typedef __attribute__((ext_vector_type(16))) __bf16 v16bf;
typedef __attribute__((ext_vector_type(8)))  float  v8f;

// Problem constants (fixed by the reference)
#define BB 4
#define SS 2048
#define DD 1024
#define HH 16
#define DEP 64
#define FF 4096
#define MM (BB*SS)   // 8192 rows

#define LDK 72       // bf16 LDS tile stride (even, 16B-aligned rows, bank-spread)
#define SLD 65       // fp32 logits tile stride (bank-spread)

static __device__ __forceinline__ unsigned short f32_to_bf16(float f) {
  unsigned int u = __float_as_uint(f);
  unsigned int r = (u + 0x7FFFu + ((u >> 16) & 1u)) >> 16;  // RNE
  return (unsigned short)r;
}

struct Frag { union { v16bf v; unsigned int u[8]; }; };
union U4 { uint4 v; unsigned short s[8]; unsigned int d[4]; };

// A-matrix 16x32 bf16 fragment from LDS tile stored row-major [m][k], stride ldk (even).
// ISA 7.12.2: lane<16 holds row=lane, K=0..7 & 16..23; lane>=16 holds row=lane-16, K=8..15 & 24..31.
static __device__ __forceinline__ Frag load_a(const unsigned short* lds, int row0,
                                              int kbase, int ldk, int lane) {
  Frag f;
  const int half = lane >> 4, r = lane & 15;
  const unsigned int* p = (const unsigned int*)lds;
  const int rowoff = (row0 + r) * ldk;
#pragma unroll
  for (int j = 0; j < 8; ++j) {
    int k = kbase + ((j < 4) ? 2 * j : 16 + 2 * (j - 4)) + 8 * half;
    f.u[j] = p[(rowoff + k) >> 1];
  }
  return f;
}

// B-matrix 32x16 bf16 fragment; LDS tile stored TRANSPOSED [n][k], stride ldk.
// lanes 0-15: N=lane, K=0..7 & 16..23 ; lanes 16-31: N=lane-16, K=8..15 & 24..31.
static __device__ __forceinline__ Frag load_b(const unsigned short* lds, int n0,
                                              int kbase, int ldk, int lane) {
  Frag f;
  const int half = lane >> 4, n = lane & 15;
  const unsigned int* p = (const unsigned int*)lds;
  const int rowoff = (n0 + n) * ldk;
#pragma unroll
  for (int j = 0; j < 8; ++j) {
    int k = kbase + ((j < 4) ? 2 * j : 16 + 2 * (j - 4)) + 8 * half;
    f.u[j] = p[(rowoff + k) >> 1];
  }
  return f;
}

// -------- fp32 [K][N] -> bf16 TRANSPOSED [N][K], tiled & coalesced both sides
__global__ void cvtT_bf16_kernel(const float* __restrict__ in,
                                 unsigned short* __restrict__ out, int K, int N) {
  __shared__ float tile[32][33];
  const int k0 = blockIdx.y * 32, n0 = blockIdx.x * 32;
  const int tx = threadIdx.x, ty = threadIdx.y;  // 32 x 8
#pragma unroll
  for (int i = 0; i < 32; i += 8)
    tile[ty + i][tx] = in[(size_t)(k0 + ty + i) * N + (n0 + tx)];
  __syncthreads();
#pragma unroll
  for (int i = 0; i < 32; i += 8)
    out[(size_t)(n0 + ty + i) * K + (k0 + tx)] = f32_to_bf16(tile[tx][ty + i]);
}

// ---------------- generic 64x64-tile WMMA GEMM ----------------
// C[M,N] = epilogue(A[M,K] * W[K,N] + bias[N]); W given pre-transposed bf16 [N][K].
// MODE 0: fp32 store [M,N]
// MODE 1: bf16 store scattered to [B,H,S,DEP], value scaled by `scale` (QKV)
// MODE 2: bf16 ReLU store [M,N]  (FFN1 -> feeds FFN2 as bf16 A)
// A_BF16: A is bf16 [M,K]; else fp32 (converted while staging in LDS).
template <int MODE, bool A_BF16>
__global__ void gemm64_wmma(const void* __restrict__ Aptr,
                            const unsigned short* __restrict__ Wt,
                            const float* __restrict__ bias,
                            void* __restrict__ Cout,
                            int M, int N, int K, float scale) {
  __shared__ unsigned short As[64 * LDK];   // [m][k]
  __shared__ unsigned short Bt[64 * LDK];   // [n][k]
  const int m0 = blockIdx.y * 64, n0 = blockIdx.x * 64;
  const int lane = threadIdx.x & 31, wave = threadIdx.x >> 5;
  const int half = lane >> 4, lr = lane & 15;
  const float* Af = (const float*)Aptr;
  const unsigned short* Ab = (const unsigned short*)Aptr;

  v8f acc[4] = {};
  for (int k0 = 0; k0 < K; k0 += 64) {
    if (k0 + 64 < K) {   // prefetch next slab (global_prefetch_b8)
      __builtin_prefetch(&Wt[(size_t)(n0 + (threadIdx.x >> 1)) * K + k0 + 64 +
                             (threadIdx.x & 1) * 32], 0, 3);
      if (!A_BF16)
        __builtin_prefetch(&Af[(size_t)(m0 + (threadIdx.x >> 1)) * K + k0 + 64 +
                               (threadIdx.x & 1) * 32], 0, 3);
    }
    if (A_BF16) {
      for (int idx = threadIdx.x; idx < 64 * 8; idx += 128) {   // 8x uint4 per row
        int r = idx >> 3, c = (idx & 7) * 8;
        *(uint4*)&As[r * LDK + c] = *(const uint4*)&Ab[(size_t)(m0 + r) * K + k0 + c];
      }
    } else {
      for (int idx = threadIdx.x; idx < 64 * 16; idx += 128) {  // 16x float4 per row
        int r = idx >> 4, c = (idx & 15) * 4;
        float4 f = *(const float4*)&Af[(size_t)(m0 + r) * K + k0 + c];
        uint2 pk;
        pk.x = (unsigned)f32_to_bf16(f.x) | ((unsigned)f32_to_bf16(f.y) << 16);
        pk.y = (unsigned)f32_to_bf16(f.z) | ((unsigned)f32_to_bf16(f.w) << 16);
        *(uint2*)&As[r * LDK + c] = pk;
      }
    }
    for (int idx = threadIdx.x; idx < 64 * 8; idx += 128) {
      int n = idx >> 3, c = (idx & 7) * 8;
      *(uint4*)&Bt[n * LDK + c] = *(const uint4*)&Wt[(size_t)(n0 + n) * K + k0 + c];
    }
    __syncthreads();
#pragma unroll
    for (int kc = 0; kc < 64; kc += 32) {
      Frag a = load_a(As, wave * 16, kc, LDK, lane);
#pragma unroll
      for (int nt = 0; nt < 4; ++nt) {
        Frag b = load_b(Bt, nt * 16, kc, LDK, lane);
        acc[nt] = __builtin_amdgcn_wmma_f32_16x16x32_bf16(
            false, a.v, false, b.v, (short)0, acc[nt], false, false);
      }
    }
    __syncthreads();
  }
  // epilogue (C layout: M = 8*half + r, N = lane&15 within each 16x16 tile)
#pragma unroll
  for (int nt = 0; nt < 4; ++nt) {
#pragma unroll
    for (int r = 0; r < 8; ++r) {
      int mm = m0 + wave * 16 + half * 8 + r;
      int nn = n0 + nt * 16 + lr;
      float c = acc[nt][r] + bias[nn];
      if (MODE == 0) {
        ((float*)Cout)[(size_t)mm * N + nn] = c;
      } else if (MODE == 1) {
        int b = mm >> 11, s = mm & (SS - 1);   // S = 2048
        int h = nn >> 6, d = nn & (DEP - 1);
        ((unsigned short*)Cout)[(((size_t)b * HH + h) * SS + s) * DEP + d] =
            f32_to_bf16(c * scale);
      } else {
        ((unsigned short*)Cout)[(size_t)mm * N + nn] = f32_to_bf16(fmaxf(c, 0.f));
      }
    }
  }
}

// ---------------- flash attention (64-query tile / workgroup) ----------------
// Q pre-scaled by 1/sqrt(64). Masking matches the reference: logits := -1e9
// where (key > query) || key_padding_mask[b][key].
__global__ void flash_attn_kernel(const unsigned short* __restrict__ Q,
                                  const unsigned short* __restrict__ Kb,
                                  const unsigned short* __restrict__ Vb,
                                  const unsigned char* __restrict__ kpm,
                                  float* __restrict__ attn) {
  __shared__ unsigned short Qs[64 * LDK];  // [q][d]
  __shared__ unsigned short Ks[64 * LDK];  // [key][d]  (== Bt for Q*K^T)
  __shared__ unsigned short Vt[64 * LDK];  // [d][key]  (== Bt for P*V)
  __shared__ unsigned short Ps[64 * LDK];  // [q][key] bf16 probabilities
  __shared__ float Ss[64 * SLD];           // logits tile
  __shared__ float mrow[64], lrow[64], arow[64];
  __shared__ unsigned char kmask[64];

  const int q0 = blockIdx.x * 64;
  const int h = blockIdx.y, b = blockIdx.z;
  const size_t base = ((size_t)b * HH + h) * SS * DEP;
  const int lane = threadIdx.x & 31, wave = threadIdx.x >> 5;
  const int half = lane >> 4, lr = lane & 15;

  for (int idx = threadIdx.x; idx < 64 * 8; idx += 128) {
    int row = idx >> 3, c = (idx & 7) * 8;
    *(uint4*)&Qs[row * LDK + c] = *(const uint4*)&Q[base + (size_t)(q0 + row) * DEP + c];
  }
  if (threadIdx.x < 64) { mrow[threadIdx.x] = -1.0e30f; lrow[threadIdx.x] = 0.f; }
  v8f o[4] = {};
  __syncthreads();

  const int ktiles = (q0 >> 6) + 1;   // causal: keys <= q0+63
  for (int kt = 0; kt < ktiles; ++kt) {
    const int k0 = kt * 64;
    for (int idx = threadIdx.x; idx < 64 * 8; idx += 128) {
      int key = idx >> 3, c = (idx & 7) * 8;
      *(uint4*)&Ks[key * LDK + c] = *(const uint4*)&Kb[base + (size_t)(k0 + key) * DEP + c];
      U4 vv; vv.v = *(const uint4*)&Vb[base + (size_t)(k0 + key) * DEP + c];
#pragma unroll
      for (int j = 0; j < 8; ++j) Vt[(c + j) * LDK + key] = vv.s[j];
    }
    if (threadIdx.x < 64) kmask[threadIdx.x] = kpm[(size_t)b * SS + k0 + threadIdx.x];
    __syncthreads();

    // S = Q * K^T (per wave: a 16x64 band -> 4 tiles x 2 k-chunks = 8 wmma)
#pragma unroll
    for (int nt = 0; nt < 4; ++nt) {
      v8f s = {};
#pragma unroll
      for (int kc = 0; kc < 64; kc += 32) {
        Frag a = load_a(Qs, wave * 16, kc, LDK, lane);
        Frag bb = load_b(Ks, nt * 16, kc, LDK, lane);
        s = __builtin_amdgcn_wmma_f32_16x16x32_bf16(
            false, a.v, false, bb.v, (short)0, s, false, false);
      }
#pragma unroll
      for (int r = 0; r < 8; ++r)
        Ss[(wave * 16 + half * 8 + r) * SLD + nt * 16 + lr] = s[r];
    }
    __syncthreads();

    // online softmax, one thread per query row
    if (threadIdx.x < 64) {
      int q = threadIdx.x, qg = q0 + q;
      float m = mrow[q];
      float rmax = m;
      for (int j = 0; j < 64; ++j) {
        bool ok = ((k0 + j) <= qg) && (kmask[j] == 0);
        float v = ok ? Ss[q * SLD + j] : -1.0e9f;
        rmax = fmaxf(rmax, v);
      }
      float alpha = __expf(m - rmax);
      float l = lrow[q] * alpha;
      for (int j = 0; j < 64; ++j) {
        bool ok = ((k0 + j) <= qg) && (kmask[j] == 0);
        float v = ok ? Ss[q * SLD + j] : -1.0e9f;
        float p = __expf(v - rmax);
        l += p;
        Ps[q * LDK + j] = f32_to_bf16(p);
      }
      mrow[q] = rmax; lrow[q] = l; arow[q] = alpha;
    }
    __syncthreads();

    // O = O*alpha + P*V
#pragma unroll
    for (int nt = 0; nt < 4; ++nt) {
#pragma unroll
      for (int r = 0; r < 8; ++r)
        o[nt][r] *= arow[wave * 16 + half * 8 + r];
#pragma unroll
      for (int kc = 0; kc < 64; kc += 32) {
        Frag a = load_a(Ps, wave * 16, kc, LDK, lane);
        Frag bb = load_b(Vt, nt * 16, kc, LDK, lane);
        o[nt] = __builtin_amdgcn_wmma_f32_16x16x32_bf16(
            false, a.v, false, bb.v, (short)0, o[nt], false, false);
      }
    }
    __syncthreads();
  }

  // normalize and scatter to [B,S,D] (undoes the head transpose)
#pragma unroll
  for (int nt = 0; nt < 4; ++nt) {
#pragma unroll
    for (int r = 0; r < 8; ++r) {
      int qrow = wave * 16 + half * 8 + r;
      float inv = 1.f / lrow[qrow];
      int s = q0 + qrow, d = nt * 16 + lr;
      attn[((size_t)b * SS + s) * DD + h * DEP + d] = o[nt][r] * inv;
    }
  }
}

// ---------------- residual add + LayerNorm (one row per block) --------------
__global__ void add_ln_kernel(const float* __restrict__ x, const float* __restrict__ y,
                              const float* __restrict__ g, const float* __restrict__ be,
                              float* __restrict__ out) {
  __shared__ float red[256];
  const int row = blockIdx.x;
  const int t = threadIdx.x;
  const float4* xr = (const float4*)(x + (size_t)row * DD);
  const float4* yr = (const float4*)(y + (size_t)row * DD);
  float4 a = xr[t], bb = yr[t];
  float v0 = a.x + bb.x, v1 = a.y + bb.y, v2 = a.z + bb.z, v3 = a.w + bb.w;
  red[t] = v0 + v1 + v2 + v3; __syncthreads();
  for (int off = 128; off > 0; off >>= 1) {
    if (t < off) red[t] += red[t + off];
    __syncthreads();
  }
  float mu = red[0] * (1.f / DD);
  __syncthreads();
  float d0 = v0 - mu, d1 = v1 - mu, d2 = v2 - mu, d3 = v3 - mu;
  red[t] = d0 * d0 + d1 * d1 + d2 * d2 + d3 * d3; __syncthreads();
  for (int off = 128; off > 0; off >>= 1) {
    if (t < off) red[t] += red[t + off];
    __syncthreads();
  }
  float rstd = rsqrtf(red[0] * (1.f / DD) + 1e-5f);
  float4 gg = ((const float4*)g)[t], ee = ((const float4*)be)[t];
  float4 o;
  o.x = d0 * rstd * gg.x + ee.x;
  o.y = d1 * rstd * gg.y + ee.y;
  o.z = d2 * rstd * gg.z + ee.z;
  o.w = d3 * rstd * gg.w + ee.w;
  ((float4*)(out + (size_t)row * DD))[t] = o;
}

// ---------------- driver ----------------
extern "C" void kernel_launch(void* const* d_in, const int* in_sizes, int n_in,
                              void* d_out, int out_size, void* d_ws, size_t ws_size,
                              hipStream_t stream) {
  (void)in_sizes; (void)n_in; (void)out_size; (void)ws_size;
  const float* src = (const float*)d_in[0];
  // d_in[1] = src_mask (tril, implied by causal loop); d_in[2] = key_padding_mask
  const unsigned char* kpm = (const unsigned char*)d_in[2];
  const float* Wq = (const float*)d_in[3];  const float* bq = (const float*)d_in[4];
  const float* Wk = (const float*)d_in[5];  const float* bk = (const float*)d_in[6];
  const float* Wv = (const float*)d_in[7];  const float* bv = (const float*)d_in[8];
  const float* Wo = (const float*)d_in[9];  const float* bo = (const float*)d_in[10];
  const float* W1 = (const float*)d_in[11]; const float* b1 = (const float*)d_in[12];
  const float* W2 = (const float*)d_in[13]; const float* b2 = (const float*)d_in[14];
  const float* g1 = (const float*)d_in[15]; const float* be1 = (const float*)d_in[16];
  const float* g2 = (const float*)d_in[17]; const float* be2 = (const float*)d_in[18];
  float* out = (float*)d_out;

  char* ws = (char*)d_ws;
  size_t off = 0;
  auto alloc = [&](size_t bytes) -> void* {
    void* p = ws + off; off += (bytes + 255) & ~(size_t)255; return p;
  };
  unsigned short* wq_bf = (unsigned short*)alloc((size_t)DD * DD * 2);
  unsigned short* wk_bf = (unsigned short*)alloc((size_t)DD * DD * 2);
  unsigned short* wv_bf = (unsigned short*)alloc((size_t)DD * DD * 2);
  unsigned short* wo_bf = (unsigned short*)alloc((size_t)DD * DD * 2);
  unsigned short* w1_bf = (unsigned short*)alloc((size_t)DD * FF * 2);
  unsigned short* w2_bf = (unsigned short*)alloc((size_t)FF * DD * 2);
  unsigned short* q_bf  = (unsigned short*)alloc((size_t)MM * DD * 2);
  unsigned short* k_bf  = (unsigned short*)alloc((size_t)MM * DD * 2);
  unsigned short* v_bf  = (unsigned short*)alloc((size_t)MM * DD * 2);
  float* attn = (float*)alloc((size_t)MM * DD * 4);
  float* oprj = (float*)alloc((size_t)MM * DD * 4);
  float* x1   = (float*)alloc((size_t)MM * DD * 4);
  unsigned short* hbf = (unsigned short*)alloc((size_t)MM * FF * 2);
  float* f2   = (float*)alloc((size_t)MM * DD * 4);

  // weights: fp32 [K][N] -> bf16 transposed [N][K] (coalesced GEMM staging)
  dim3 tblk(32, 8);
  cvtT_bf16_kernel<<<dim3(DD / 32, DD / 32), tblk, 0, stream>>>(Wq, wq_bf, DD, DD);
  cvtT_bf16_kernel<<<dim3(DD / 32, DD / 32), tblk, 0, stream>>>(Wk, wk_bf, DD, DD);
  cvtT_bf16_kernel<<<dim3(DD / 32, DD / 32), tblk, 0, stream>>>(Wv, wv_bf, DD, DD);
  cvtT_bf16_kernel<<<dim3(DD / 32, DD / 32), tblk, 0, stream>>>(Wo, wo_bf, DD, DD);
  cvtT_bf16_kernel<<<dim3(FF / 32, DD / 32), tblk, 0, stream>>>(W1, w1_bf, DD, FF);
  cvtT_bf16_kernel<<<dim3(DD / 32, FF / 32), tblk, 0, stream>>>(W2, w2_bf, FF, DD);

  dim3 blk(128);
  // QKV projections -> bf16 [B,H,S,64]; Q pre-scaled by 1/sqrt(64)
  dim3 gProj(DD / 64, MM / 64);
  gemm64_wmma<1, false><<<gProj, blk, 0, stream>>>(src, wq_bf, bq, q_bf, MM, DD, DD, 0.125f);
  gemm64_wmma<1, false><<<gProj, blk, 0, stream>>>(src, wk_bf, bk, k_bf, MM, DD, DD, 1.0f);
  gemm64_wmma<1, false><<<gProj, blk, 0, stream>>>(src, wv_bf, bv, v_bf, MM, DD, DD, 1.0f);

  // flash attention -> attn [B,S,D] fp32
  flash_attn_kernel<<<dim3(SS / 64, HH, BB), blk, 0, stream>>>(q_bf, k_bf, v_bf, kpm, attn);

  // output projection
  gemm64_wmma<0, false><<<gProj, blk, 0, stream>>>(attn, wo_bf, bo, oprj, MM, DD, DD, 1.0f);
  // x1 = LN(src + attn_out)
  add_ln_kernel<<<MM, 256, 0, stream>>>(src, oprj, g1, be1, x1);

  // FFN
  dim3 gF1(FF / 64, MM / 64);
  gemm64_wmma<2, false><<<gF1, blk, 0, stream>>>(x1, w1_bf, b1, hbf, MM, FF, DD, 1.0f);
  gemm64_wmma<0, true><<<gProj, blk, 0, stream>>>(hbf, w2_bf, b2, f2, MM, DD, FF, 1.0f);

  // out = LN(x1 + ffn)
  add_ln_kernel<<<MM, 256, 0, stream>>>(x1, f2, g2, be2, out);
}
